// LightningTransformer_62758062129469
// MI455X (gfx1250) — matrix operand order repaired and checked by reading, not compiled, not run on hardware
//
#include <hip/hip_runtime.h>
#include <hip/hip_bf16.h>

// ---------------- constants (match reference) ----------------
constexpr int kB = 4, kH = 8, kS = 2048, kHD = 80;
constexpr int kFF = 480;            // MULT * HD
constexpr int kChunk = 128, kNC = kS / kChunk;
constexpr int kDP = 96;             // HD padded to multiple of 32 for bf16 WMMA K-dim
constexpr long kRows = (long)kB * kH * kS;   // 65536 rows of HD

typedef __attribute__((ext_vector_type(16))) __bf16 v16bf;
typedef __attribute__((ext_vector_type(8)))  __bf16 v8bf;
typedef __attribute__((ext_vector_type(8)))  float  v8f;

__device__ __forceinline__ v8f wmma_bf16(v16bf a, v16bf b, v8f c) {
    return __builtin_amdgcn_wmma_f32_16x16x32_bf16(
        false, a, false, b, (short)0, c, false, false);
}

// A fragment, 16x32 bf16, row-major source (tile origin p, stride ld elems).
// lane = M + 16*hi ; per lane the needed K values are two contiguous 16B runs.
__device__ __forceinline__ v16bf ldsA(const __bf16* p, int ld) {
    int lane = threadIdx.x & 31;
    const __bf16* r = p + (lane & 15) * ld + (((lane >> 4) & 1) << 3);
    v8bf lo = *(const v8bf*)(r);        // K = hb+0..7
    v8bf hi = *(const v8bf*)(r + 16);   // K = 16+hb+0..7
    return __builtin_shufflevector(lo, hi, 0,1,2,3,4,5,6,7,8,9,10,11,12,13,14,15);
}

// B fragment, 32x16 bf16, row-major K x N source: lane = N col + 16*hi, K in table order (8hi+0..7, 16+8hi+0..7).
__device__ __forceinline__ v16bf ldsB(const __bf16* p, int ld) {
    int lane = threadIdx.x & 31;
    const __bf16* c = p + (lane & 15) + ((lane >> 4) & 1) * 8 * ld;
    v16bf b;
#pragma unroll
    for (int s = 0; s < 8; ++s) {
        b[s]     = c[s * ld];
        b[s + 8] = c[(16 + s) * ld];
    }
    return b;
}

// ---------------- weight convert / pad to bf16 ----------------
__global__ void __launch_bounds__(256)
cvt_w_kernel(const float* __restrict__ W1, const float* __restrict__ W2,
             __bf16* __restrict__ W1p, __bf16* __restrict__ W2p) {
    int i = blockIdx.x * 256 + threadIdx.x;
    if (i < kDP * kFF) {                 // W1p: 96 x 480, zero-pad K 80..95
        int k = i / kFF, n = i % kFF;
        W1p[i] = (k < kHD) ? (__bf16)W1[k * kFF + n] : (__bf16)0.f;
    }
    if (i < kFF * kHD) {                 // W2p: 480 x 80
        W2p[i] = (__bf16)W2[i];
    }
}

// ---------------- fused (add +) LayerNorm over rows of 80 ----------------
__global__ void __launch_bounds__(256)
ln_kernel(const float* __restrict__ xin, const float* __restrict__ addin,
          const float* __restrict__ gamma, const float* __restrict__ beta,
          float* __restrict__ out_f32, __bf16* __restrict__ out_bf16, long nrows) {
    const int lane = threadIdx.x & 31;
    const long row = (long)blockIdx.x * 8 + (threadIdx.x >> 5);
    if (row >= nrows) return;
    const float* xr = xin + row * kHD;
    const float* ar = addin ? (addin + row * kHD) : nullptr;

    float x0 = xr[lane]      + (ar ? ar[lane]      : 0.f);
    float x1 = xr[lane + 32] + (ar ? ar[lane + 32] : 0.f);
    float x2 = (lane < 16) ? (xr[lane + 64] + (ar ? ar[lane + 64] : 0.f)) : 0.f;

    float sum = x0 + x1 + x2;
    float sq  = x0*x0 + x1*x1 + x2*x2;
#pragma unroll
    for (int o = 16; o > 0; o >>= 1) {
        sum += __shfl_xor(sum, o, 32);
        sq  += __shfl_xor(sq,  o, 32);
    }
    float mean = sum * (1.f / kHD);
    float var  = sq * (1.f / kHD) - mean * mean;
    float rst  = rsqrtf(var + 1e-5f);

    float y0 = (x0 - mean) * rst * gamma[lane]      + beta[lane];
    float y1 = (x1 - mean) * rst * gamma[lane + 32] + beta[lane + 32];
    float y2 = (lane < 16) ? ((x2 - mean) * rst * gamma[lane + 64] + beta[lane + 64]) : 0.f;

    if (out_f32) {
        out_f32[row * kHD + lane]      = y0;
        out_f32[row * kHD + lane + 32] = y1;
        if (lane < 16) out_f32[row * kHD + lane + 64] = y2;
    }
    if (out_bf16) {
        out_bf16[row * kHD + lane]      = (__bf16)y0;
        out_bf16[row * kHD + lane + 32] = (__bf16)y1;
        if (lane < 16) out_bf16[row * kHD + lane + 64] = (__bf16)y2;
    }
}

// ---------------- attention pass 1: U_c = (K*kdec)^T V per (b,h,chunk) ----------------
__global__ void __launch_bounds__(256)
kvu_kernel(const __bf16* __restrict__ x, const float* __restrict__ slopes,
           float* __restrict__ U) {
    __shared__ __bf16 Xb[kChunk * kDP];    // V, row-major (i x d)
    __shared__ __bf16 KdT[kDP * kChunk];   // (K * kdec)^T, row-major (d x i)
    __shared__ float  kdec[kChunk];

    const int bh = blockIdx.x / kNC, c = blockIdx.x % kNC;
    const int h  = bh % kH;
    const float s = slopes[h];
    const __bf16* cbase = x + ((size_t)bh * kS + (size_t)c * kChunk) * kHD;
    const int tid = threadIdx.x, wave = tid >> 5, lane = tid & 31;

    for (int i = tid; i < kChunk; i += 256) kdec[i] = __expf(-s * (float)(kChunk - i));
    __syncthreads();

    for (int idx = tid; idx < kChunk * kDP; idx += 256) {
        int r = idx / kDP, cc = idx % kDP;
        __bf16 val = (cc < kHD) ? cbase[r * kHD + cc] : (__bf16)0.f;
        Xb[idx] = val;
        KdT[cc * kChunk + r] = (__bf16)((float)val * kdec[r]);
    }
    __syncthreads();

    float* Ub = U + ((size_t)bh * kNC + c) * kHD * kHD;
    for (int t = wave; t < 25; t += 8) {                 // 5x5 tiles (d,e < 80)
        int td = t / 5, te = t % 5;
        v8f acc = {};
#pragma unroll
        for (int ks = 0; ks < 4; ++ks) {
            v16bf a = ldsA(KdT + td * 16 * kChunk + ks * 32, kChunk);
            v16bf b = ldsB(Xb + (ks * 32) * kDP + te * 16, kDP);
            acc = wmma_bf16(a, b, acc);
        }
        int e = te * 16 + (lane & 15), hi = lane >> 4;
#pragma unroll
        for (int r = 0; r < 8; ++r)
            Ub[(td * 16 + r + 8 * hi) * kHD + e] = acc[r];
    }
}

// ---------------- attention pass 2: tiny linear scan of kv states ----------------
__global__ void __launch_bounds__(256)
scan_kernel(const float* __restrict__ U, const float* __restrict__ slopes,
            __bf16* __restrict__ KV) {
    const int bh = blockIdx.x, h = bh % kH, tid = threadIdx.x;
    const float blk = __expf(-slopes[h] * (float)kChunk);
    float st[25];                                   // 6400 = 25*256 elems of state
#pragma unroll
    for (int j = 0; j < 25; ++j) st[j] = 0.f;

    const float* Ub = U + (size_t)bh * kNC * kHD * kHD;
    __bf16* KVb = KV + (size_t)bh * kNC * kDP * kDP;

    for (int c = 0; c < kNC; ++c) {
        __bf16* kvc = KVb + (size_t)c * kDP * kDP;
        // write pre-chunk state, padded to 96x96 with zeros
#pragma unroll
        for (int j = 0; j < 25; ++j) {
            int idx = j * 256 + tid;
            int d = idx / kHD, e = idx % kHD;
            kvc[d * kDP + e] = (__bf16)st[j];
        }
        for (int idx = tid; idx < kDP * kDP; idx += 256) {
            int d = idx / kDP, e = idx % kDP;
            if (d >= kHD || e >= kHD) kvc[idx] = (__bf16)0.f;
        }
        // advance: kv <- blk*kv + U_c
#pragma unroll
        for (int j = 0; j < 25; ++j)
            st[j] = blk * st[j] + Ub[(size_t)c * kHD * kHD + j * 256 + tid];
    }
}

// ---------------- attention pass 3: O = mask(QK^T)V + qdec*(Q kv) ----------------
__global__ void __launch_bounds__(256)
attn_o_kernel(const __bf16* __restrict__ x, const float* __restrict__ slopes,
              const __bf16* __restrict__ KV, float* __restrict__ out) {
    extern __shared__ char smem[];
    __bf16* Xb  = (__bf16*)smem;              // 128 x 96 (serves as Q and V)
    __bf16* KbT = Xb + kChunk * kDP;          //  96 x 128 (K^T)
    __bf16* Sb  = KbT + kDP * kChunk;         // 128 x 128 masked scores (bf16)
    float* qdec = (float*)(Sb + kChunk * kChunk);  // exp(-s*i), 128
    float* einv = qdec + kChunk;                   // exp(+s*j), 128

    const int bh = blockIdx.x / kNC, c = blockIdx.x % kNC;
    const int h  = bh % kH;
    const float s = slopes[h];
    const __bf16* cbase = x + ((size_t)bh * kS + (size_t)c * kChunk) * kHD;
    const __bf16* kvc   = KV + ((size_t)bh * kNC + c) * kDP * kDP;
    float* obase = out + ((size_t)bh * kS + (size_t)c * kChunk) * kHD;
    const int tid = threadIdx.x, wave = tid >> 5, lane = tid & 31;

    for (int i = tid; i < kChunk; i += 256) {
        qdec[i] = __expf(-s * (float)i);
        einv[i] = __expf( s * (float)i);
    }
    for (int idx = tid; idx < kChunk * kDP; idx += 256) {
        int r = idx / kDP, cc = idx % kDP;
        __bf16 val = (cc < kHD) ? cbase[r * kHD + cc] : (__bf16)0.f;
        Xb[idx] = val;
        KbT[cc * kChunk + r] = val;
    }
    __syncthreads();

    // S = Q K^T with causal decay mask: 8x8 tiles
    for (int t = wave; t < 64; t += 8) {
        int tm = t >> 3, tn = t & 7;
        v8f acc = {};
#pragma unroll
        for (int ks = 0; ks < 3; ++ks) {
            v16bf a = ldsA(Xb + tm * 16 * kDP + ks * 32, kDP);
            v16bf b = ldsB(KbT + (ks * 32) * kChunk + tn * 16, kChunk);
            acc = wmma_bf16(a, b, acc);
        }
        int n = tn * 16 + (lane & 15), hi = lane >> 4;
        float ej = einv[n];
#pragma unroll
        for (int r = 0; r < 8; ++r) {
            int i = tm * 16 + r + 8 * hi;
            float v = (i >= n) ? acc[r] * qdec[i] * ej : 0.f;  // = acc*exp(-s*(i-n))
            Sb[i * kChunk + n] = (__bf16)v;
        }
    }
    __syncthreads();

    // O = S V + qdec[i] * (Q kv): 8x5 tiles (N < 80)
    for (int t = wave; t < 40; t += 8) {
        int tm = t / 5, tn = t % 5;
        v8f acc = {};
#pragma unroll
        for (int ks = 0; ks < 4; ++ks) {              // intra-chunk
            v16bf a = ldsA(Sb + tm * 16 * kChunk + ks * 32, kChunk);
            v16bf b = ldsB(Xb + (ks * 32) * kDP + tn * 16, kDP);
            acc = wmma_bf16(a, b, acc);
        }
        v8f acc2 = {};
#pragma unroll
        for (int ks = 0; ks < 3; ++ks) {              // inter-chunk (kv state, global/L2)
            v16bf a = ldsA(Xb + tm * 16 * kDP + ks * 32, kDP);
            v16bf b = ldsB(kvc + (ks * 32) * kDP + tn * 16, kDP);
            acc2 = wmma_bf16(a, b, acc2);
        }
        int n = tn * 16 + (lane & 15), hi = lane >> 4;
#pragma unroll
        for (int r = 0; r < 8; ++r) {
            int i = tm * 16 + r + 8 * hi;
            obase[(size_t)i * kHD + n] = acc[r] + qdec[i] * acc2[r];
        }
    }
}

// ---------------- fused FFN: leaky_relu(X@W1+b1)@W2+b2, 32 rows per block ----------------
__global__ void __launch_bounds__(256)
ffn_kernel(const __bf16* __restrict__ x,
           const __bf16* __restrict__ W1p, const float* __restrict__ b1,
           const __bf16* __restrict__ W2p, const float* __restrict__ b2,
           float* __restrict__ out) {
    __shared__ __bf16 Xs[32 * kDP];
    __shared__ __bf16 Mid[32 * kFF];
    const int tid = threadIdx.x, wave = tid >> 5, lane = tid & 31;
    const size_t row0 = (size_t)blockIdx.x * 32;

    for (int i = tid; i < 32 * kDP; i += 256) {
        int r = i / kDP, c = i % kDP;
        Xs[i] = (c < kHD) ? x[(row0 + r) * kHD + c] : (__bf16)0.f;
    }
    __syncthreads();

    for (int t = wave; t < 60; t += 8) {              // (32x96)@(96x480): 2x30 tiles
        int tm = t / 30, tn = t % 30;
        v8f acc = {};
#pragma unroll
        for (int ks = 0; ks < 3; ++ks) {
            v16bf a = ldsA(Xs + tm * 16 * kDP + ks * 32, kDP);
            v16bf b = ldsB(W1p + (ks * 32) * kFF + tn * 16, kFF);
            acc = wmma_bf16(a, b, acc);
        }
        int n = tn * 16 + (lane & 15), hi = lane >> 4;
        float bias = b1[n];
#pragma unroll
        for (int r = 0; r < 8; ++r) {
            float v = acc[r] + bias;
            v = (v >= 0.f) ? v : 0.01f * v;           // leaky_relu(0.01)
            Mid[(tm * 16 + r + 8 * hi) * kFF + n] = (__bf16)v;
        }
    }
    __syncthreads();

    for (int t = wave; t < 10; t += 8) {              // (32x480)@(480x80): 2x5 tiles
        int tm = t / 5, tn = t % 5;
        v8f acc = {};
#pragma unroll
        for (int ks = 0; ks < 15; ++ks) {
            v16bf a = ldsA(Mid + tm * 16 * kFF + ks * 32, kFF);
            v16bf b = ldsB(W2p + (ks * 32) * kHD + tn * 16, kHD);
            acc = wmma_bf16(a, b, acc);
        }
        int n = tn * 16 + (lane & 15), hi = lane >> 4;
        float bias = b2[n];
#pragma unroll
        for (int r = 0; r < 8; ++r)
            out[(row0 + tm * 16 + r + 8 * hi) * kHD + n] = acc[r] + bias;
    }
}

// ---------------- mean over heads ----------------
__global__ void __launch_bounds__(256)
mean_kernel(const float* __restrict__ xin, float* __restrict__ out) {
    long i = (long)blockIdx.x * 256 + threadIdx.x;    // over B*S*HD
    if (i >= (long)kB * kS * kHD) return;
    int  d = (int)(i % kHD);
    long t = i / kHD;
    int  ss = (int)(t % kS);
    int  b  = (int)(t / kS);
    float acc = 0.f;
#pragma unroll
    for (int h = 0; h < kH; ++h)
        acc += xin[(((long)b * kH + h) * kS + ss) * kHD + d];
    out[i] = acc * (1.f / kH);
}

// ---------------- host-side orchestration ----------------
extern "C" void kernel_launch(void* const* d_in, const int* in_sizes, int n_in,
                              void* d_out, int out_size, void* d_ws, size_t ws_size,
                              hipStream_t stream) {
    (void)in_sizes; (void)n_in; (void)out_size; (void)ws_size;
    const float* x      = (const float*)d_in[0];
    const float* W1     = (const float*)d_in[1];
    const float* b1     = (const float*)d_in[2];
    const float* W2     = (const float*)d_in[3];
    const float* b2     = (const float*)d_in[4];
    const float* gamma  = (const float*)d_in[5];
    const float* beta   = (const float*)d_in[6];
    const float* slopes = (const float*)d_in[7];

    char* ws = (char*)d_ws;
    const size_t fbytes = (size_t)kRows * kHD * sizeof(float);      // 20.97 MB
    const size_t hbytes = (size_t)kRows * kHD * sizeof(__bf16);     // 10.49 MB
    const size_t ubytes = (size_t)kB * kH * kNC * kHD * kHD * sizeof(float);   // 13.1 MB
    const size_t kvby   = (size_t)kB * kH * kNC * kDP * kDP * sizeof(__bf16);  // 9.4 MB
    float*  resid = (float*)ws;
    float*  curf  = (float*)(ws + fbytes);
    float*  tmpf  = (float*)(ws + 2 * fbytes);
    __bf16* curb  = (__bf16*)(ws + 3 * fbytes);
    float*  Ubuf  = (float*)(ws + 3 * fbytes + hbytes);
    __bf16* KVbuf = (__bf16*)(ws + 3 * fbytes + hbytes + ubytes);
    __bf16* W1p   = (__bf16*)(ws + 3 * fbytes + hbytes + ubytes + kvby);
    __bf16* W2p   = (__bf16*)((char*)W1p + (size_t)kDP * kFF * sizeof(__bf16));

    const int lnBlocks   = (int)(kRows / 8);
    const int ffnBlocks  = (int)(kRows / 32);
    const int chBlocks   = kB * kH * kNC;            // 512
    const int attn3Smem  = (kChunk * kDP + kDP * kChunk + kChunk * kChunk) * 2
                         + 2 * kChunk * 4;           // 82,944 B dynamic LDS

    cvt_w_kernel<<<(kDP * kFF + 255) / 256, 256, 0, stream>>>(W1, W2, W1p, W2p);

    // hdn = LN(x); residual kept f32; bf16 copy feeds matrix kernels
    ln_kernel<<<lnBlocks, 256, 0, stream>>>(x, nullptr, gamma, beta, resid, curb, kRows);

    for (int layer = 0; layer < 2; ++layer) {
        kvu_kernel<<<chBlocks, 256, 0, stream>>>(curb, slopes, Ubuf);
        scan_kernel<<<kB * kH, 256, 0, stream>>>(Ubuf, slopes, KVbuf);
        attn_o_kernel<<<chBlocks, 256, attn3Smem, stream>>>(curb, slopes, KVbuf, tmpf);
        ln_kernel<<<lnBlocks, 256, 0, stream>>>(tmpf, resid, gamma, beta, nullptr, curb, kRows);
        ffn_kernel<<<ffnBlocks, 256, 0, stream>>>(curb, W1p, b1, W2p, b2, tmpf);
        ln_kernel<<<lnBlocks, 256, 0, stream>>>(tmpf, resid, gamma, beta, curf, curb, kRows);
    }

    mean_kernel<<<(int)((kB * (long)kS * kHD + 255) / 256), 256, 0, stream>>>(
        curf, (float*)d_out);
}
